// PointNet2Autoencoder_85993835200544
// MI455X (gfx1250) — compile-verified, matrix operand order checked by reading, not run on hardware
//
#include <hip/hip_runtime.h>

// ---------------------------------------------------------------------------
// PointNet++ autoencoder for MI455X (gfx1250, wave32).
// Dense math via V_WMMA_F32_16X16X32_F16 (f16 in, f32 accum), 32x32 output
// macro-tile per wave (2x2 WMMA tiles) for 2x operand reuse: 8 b128 loads
// feed 4 WMMAs per K-step instead of 4 loads per 1 WMMA.
// Decoder layer-1 exploits input structure: inp = [broadcast(z), grid] so
//   h1 = relu(bn(z@Wz^T (64x1024x512 GEMM) + grid@Wg^T (tiny) + bias))
// instead of a 131072x1026x512 GEMM.
// ---------------------------------------------------------------------------

typedef __attribute__((ext_vector_type(16))) _Float16 v16h;
typedef __attribute__((ext_vector_type(8)))  _Float16 v8h;
typedef __attribute__((ext_vector_type(8)))  float    v8f;

#define CDIV(a,b) (((a)+(b)-1)/(b))

// ------------------------- farthest point sampling -------------------------
__global__ void fps_kernel(const float* __restrict__ xyz, int N, int npoint,
                           int* __restrict__ outIdx, float* __restrict__ outXyz)
{
  __shared__ float dist[2048];
  __shared__ float rv[256];
  __shared__ int   ri[256];
  __shared__ float c[3];
  const int b = blockIdx.x, tid = threadIdx.x;
  const float* pts = xyz + (size_t)b * N * 3;
  for (int i = tid; i < N; i += blockDim.x) dist[i] = 3.402823e38f;
  __syncthreads();
  int far = 0;
  for (int j = 0; j < npoint; ++j) {
    if (tid == 0) {
      outIdx[b * npoint + j] = far;
      c[0] = pts[far*3+0]; c[1] = pts[far*3+1]; c[2] = pts[far*3+2];
      float* o = outXyz + ((size_t)b * npoint + j) * 3;
      o[0] = c[0]; o[1] = c[1]; o[2] = c[2];
    }
    __syncthreads();
    float cx = c[0], cy = c[1], cz = c[2];
    float bv = -1.f; int bi = 0x7fffffff;
    for (int i = tid; i < N; i += blockDim.x) {
      float dx = pts[i*3+0]-cx, dy = pts[i*3+1]-cy, dz = pts[i*3+2]-cz;
      float d = dx*dx + dy*dy + dz*dz;
      float nd = fminf(dist[i], d);
      dist[i] = nd;
      if (nd > bv) { bv = nd; bi = i; }   // earliest max within thread
    }
    rv[tid] = bv; ri[tid] = bi;
    __syncthreads();
    for (int s = blockDim.x >> 1; s > 0; s >>= 1) {
      if (tid < s) {
        if (rv[tid+s] > rv[tid] || (rv[tid+s] == rv[tid] && ri[tid+s] < ri[tid])) {
          rv[tid] = rv[tid+s]; ri[tid] = ri[tid+s];
        }
      }
      __syncthreads();
    }
    far = ri[0];
    __syncthreads();
  }
}

// ------------------- ball query: first K indices within r2 -----------------
// one wave32 per (b,s); ballot-based compaction keeps scan order (== sorted)
__global__ void ballq_kernel(const float* __restrict__ xyz, const float* __restrict__ nxyz,
                             int N, int S, int K, float r2, int* __restrict__ idxOut)
{
  const int bs = blockIdx.x;
  const int b  = bs / S;
  const int lane = threadIdx.x;
  const float* pts = xyz + (size_t)b * N * 3;
  const float cx = nxyz[(size_t)bs*3+0], cy = nxyz[(size_t)bs*3+1], cz = nxyz[(size_t)bs*3+2];
  int* out = idxOut + (size_t)bs * K;
  int count = 0;
  for (int base = 0; base < N && count < K; base += 32) {
    int i = base + lane;
    bool q = false;
    if (i < N) {
      float dx = pts[i*3+0]-cx, dy = pts[i*3+1]-cy, dz = pts[i*3+2]-cz;
      q = (dx*dx + dy*dy + dz*dz) <= r2;
    }
    unsigned mask = (unsigned)__ballot(q);            // wave32: low 32 bits
    int pos = count + __popc(mask & ((1u << lane) - 1u));
    if (q && pos < K) out[pos] = i;
    count += __popc(mask);
  }
  if (count > K) count = K;
  __syncthreads();
  int first = (count > 0) ? out[0] : 0;
  for (int slot = count + lane; slot < K; slot += 32) out[slot] = first;
}

// --------------------- grouped-input builders (f16, padded K) --------------
__global__ void group1_kernel(const float* __restrict__ xyz, const float* __restrict__ nx,
                              const int* __restrict__ idx, _Float16* __restrict__ X)
{
  int row = blockIdx.x * blockDim.x + threadIdx.x;      // (b*64+s)*32+k
  if (row >= 64*64*32) return;
  int bs = row >> 5, b = bs >> 6;
  int j = idx[row];
  const float* p = xyz + ((size_t)b*2048 + j) * 3;
  const float* c = nx  + (size_t)bs * 3;
  _Float16* o = X + (size_t)row * 32;
  o[0] = (_Float16)(p[0]-c[0]); o[1] = (_Float16)(p[1]-c[1]); o[2] = (_Float16)(p[2]-c[2]);
#pragma unroll
  for (int t = 3; t < 32; ++t) o[t] = (_Float16)0.f;
}

__global__ void group2_kernel(const float* __restrict__ nx1, const float* __restrict__ nx2,
                              const float* __restrict__ feats1, const int* __restrict__ idx,
                              _Float16* __restrict__ X)
{
  int row = blockIdx.x * blockDim.x + threadIdx.x;      // (b*16+s)*32+k
  if (row >= 64*16*32) return;
  int bs = row >> 5, b = bs >> 4;
  int j = idx[row];                                     // index into 64 SA1 centroids
  const float* p = nx1 + ((size_t)b*64 + j) * 3;
  const float* c = nx2 + (size_t)bs * 3;
  const float* f = feats1 + ((size_t)b*64 + j) * 128;
  _Float16* o = X + (size_t)row * 160;
  o[0] = (_Float16)(p[0]-c[0]); o[1] = (_Float16)(p[1]-c[1]); o[2] = (_Float16)(p[2]-c[2]);
  for (int t = 0; t < 128; ++t) o[3+t] = (_Float16)f[t];
  for (int t = 131; t < 160; ++t) o[t] = (_Float16)0.f;
}

__global__ void group3_kernel(const float* __restrict__ nx2, const float* __restrict__ feats2,
                              _Float16* __restrict__ X)
{
  int row = blockIdx.x * blockDim.x + threadIdx.x;      // b*16+k
  if (row >= 64*16) return;
  const float* p = nx2 + (size_t)row * 3;               // absolute xyz (global SA stage)
  const float* f = feats2 + (size_t)row * 256;
  _Float16* o = X + (size_t)row * 288;
  o[0] = (_Float16)p[0]; o[1] = (_Float16)p[1]; o[2] = (_Float16)p[2];
  for (int t = 0; t < 256; ++t) o[3+t] = (_Float16)f[t];
  for (int t = 259; t < 288; ++t) o[t] = (_Float16)0.f;
}

// -------- pack fp32 weights -> padded f16 (cout_pad x kpad), fold BN -------
__global__ void prep_layer(const float* __restrict__ W, const float* __restrict__ bias,
                           const float* __restrict__ gamma, const float* __restrict__ beta,
                           const float* __restrict__ mean, const float* __restrict__ var,
                           int cout, int cin, int srcLd, int kpad, int coutPad,
                           _Float16* __restrict__ Wh, float* __restrict__ scale,
                           float* __restrict__ shift)
{
  int o = blockIdx.x;
  if (o >= coutPad) return;
  for (int k = threadIdx.x; k < kpad; k += blockDim.x) {
    float w = (o < cout && k < cin) ? W[(size_t)o * srcLd + k] : 0.f;
    Wh[(size_t)o * kpad + k] = (_Float16)w;
  }
  if (threadIdx.x == 0) {
    float s = 0.f, sb = 0.f;
    if (o < cout) {
      if (gamma) {
        float is = rsqrtf(var[o] + 1e-5f) * gamma[o];
        s = is; sb = (bias[o] - mean[o]) * is + beta[o];
      } else { s = 1.f; sb = bias[o]; }
    }
    scale[o] = s; shift[o] = sb;
  }
}

// ---------------- WMMA helpers: per-lane operand gather --------------------
// CDNA5 ISA 7.12.2 f16 A/B layout: lane-group g=lane>>4; element e maps to
// k = e + 8*((e>>3)+g). Two contiguous 8-half runs -> two b128 loads.
__device__ __forceinline__ v16h load_tile_row(const _Float16* rowp, int kk, int g)
{
  v8h lo = *(const v8h*)(rowp + kk + 8*g);
  v8h hi = *(const v8h*)(rowp + kk + 16 + 8*g);
  v16h t;
#pragma unroll
  for (int e = 0; e < 8; ++e) { t[e] = lo[e]; t[e+8] = hi[e]; }
  return t;
}

__device__ __forceinline__ void store_tile(v8f acc, int tm, int n, int g,
                                           float s, float sb, int useAffine, int relu,
                                           int Ncols, _Float16* Yh, float* Yf, int ldy)
{
#pragma unroll
  for (int v = 0; v < 8; ++v) {
    int m = tm * 16 + v + 8 * g;                       // D layout: VGPR v -> row v+8g
    float y = acc[v];
    if (useAffine) y = y * s + sb;
    if (relu)      y = fmaxf(y, 0.f);
    if (n < Ncols) {
      if (Yf) Yf[(size_t)m * ldy + n] = y;
      else    Yh[(size_t)m * ldy + n] = (_Float16)y;
    }
  }
}

// ---------------- WMMA GEMM + BN + ReLU, 32x32 macro-tile per wave ---------
// Y[m,n] = act( (sum_k X[m,k]*Wt[n,k]) * scale[n] + shift[n] )
// X: (tiles2M*32 x K) f16, Wt: (tiles2N*32 x K) f16.
// 2 A-tiles x 2 B-tiles -> 4 accumulators; 8 b128 loads feed 4 WMMAs per K-step.
__global__ void wmma_gemm2x2(const _Float16* __restrict__ X, int K,
                             const _Float16* __restrict__ Wt,
                             const float* __restrict__ scale, const float* __restrict__ shift,
                             int tiles2M, int tiles2N, int Ncols,
                             _Float16* __restrict__ Yh, float* __restrict__ Yf,
                             int ldy, int relu)
{
  int tile = blockIdx.x * (blockDim.x >> 5) + (threadIdx.x >> 5);
  if (tile >= tiles2M * tiles2N) return;               // wave-uniform exit (EXEC stays all-1)
  const int tm = (tile / tiles2N) * 2, tn = (tile % tiles2N) * 2;
  const int lane = threadIdx.x & 31;
  const int g = lane >> 4, r = lane & 15;
  const _Float16* arow0 = X  + (size_t)(tm * 16 + r) * K;
  const _Float16* arow1 = arow0 + (size_t)16 * K;
  const _Float16* brow0 = Wt + (size_t)(tn * 16 + r) * K;
  const _Float16* brow1 = brow0 + (size_t)16 * K;
  v8f c00 = {0.f,0.f,0.f,0.f,0.f,0.f,0.f,0.f};
  v8f c01 = c00, c10 = c00, c11 = c00;
  for (int kk = 0; kk < K; kk += 32) {
    v16h a0 = load_tile_row(arow0, kk, g);
    v16h a1 = load_tile_row(arow1, kk, g);
    v16h b0 = load_tile_row(brow0, kk, g);
    v16h b1 = load_tile_row(brow1, kk, g);
    c00 = __builtin_amdgcn_wmma_f32_16x16x32_f16(false, a0, false, b0, (short)0, c00, false, false);
    c01 = __builtin_amdgcn_wmma_f32_16x16x32_f16(false, a0, false, b1, (short)0, c01, false, false);
    c10 = __builtin_amdgcn_wmma_f32_16x16x32_f16(false, a1, false, b0, (short)0, c10, false, false);
    c11 = __builtin_amdgcn_wmma_f32_16x16x32_f16(false, a1, false, b1, (short)0, c11, false, false);
  }
  const int n0 = tn * 16 + r, n1 = n0 + 16;
  float s0 = 1.f, sb0 = 0.f, s1 = 1.f, sb1 = 0.f;
  const int useAffine = scale != nullptr;
  if (useAffine) { s0 = scale[n0]; sb0 = shift[n0]; s1 = scale[n1]; sb1 = shift[n1]; }
  store_tile(c00, tm,     n0, g, s0, sb0, useAffine, relu, Ncols, Yh, Yf, ldy);
  store_tile(c01, tm,     n1, g, s1, sb1, useAffine, relu, Ncols, Yh, Yf, ldy);
  store_tile(c10, tm + 1, n0, g, s0, sb0, useAffine, relu, Ncols, Yh, Yf, ldy);
  store_tile(c11, tm + 1, n1, g, s1, sb1, useAffine, relu, Ncols, Yh, Yf, ldy);
}

// ---------------- WMMA GEMM, single 16x16 tile (odd tile counts) -----------
__global__ void wmma_gemm(const _Float16* __restrict__ X, int K,
                          const _Float16* __restrict__ Wt,
                          const float* __restrict__ scale, const float* __restrict__ shift,
                          int tilesM, int tilesN, int Ncols,
                          _Float16* __restrict__ Yh, float* __restrict__ Yf,
                          int ldy, int relu)
{
  int tile = blockIdx.x * (blockDim.x >> 5) + (threadIdx.x >> 5);
  if (tile >= tilesM * tilesN) return;
  const int tm = tile / tilesN, tn = tile % tilesN;
  const int lane = threadIdx.x & 31;
  const int g = lane >> 4, r = lane & 15;
  const _Float16* arow = X  + (size_t)(tm * 16 + r) * K;
  const _Float16* brow = Wt + (size_t)(tn * 16 + r) * K;
  v8f acc = {0.f,0.f,0.f,0.f,0.f,0.f,0.f,0.f};
  for (int kk = 0; kk < K; kk += 32) {
    v16h a = load_tile_row(arow, kk, g);
    v16h b = load_tile_row(brow, kk, g);
    acc = __builtin_amdgcn_wmma_f32_16x16x32_f16(false, a, false, b, (short)0, acc, false, false);
  }
  const int n = tn * 16 + r;
  float s = 1.f, sb = 0.f;
  const int useAffine = scale != nullptr;
  if (useAffine) { s = scale[n]; sb = shift[n]; }
  store_tile(acc, tm, n, g, s, sb, useAffine, relu, Ncols, Yh, Yf, ldy);
}

// ------------------------------ max over group -----------------------------
__global__ void maxpool_kernel(const _Float16* __restrict__ Y, int ld, int Kk, int C, int G,
                               float* __restrict__ out)
{
  int t = blockIdx.x * blockDim.x + threadIdx.x;
  if (t >= G * C) return;
  int g = t / C, c = t % C;
  const _Float16* p = Y + (size_t)g * Kk * ld + c;
  float m = (float)p[0];
  for (int k = 1; k < Kk; ++k) m = fmaxf(m, (float)p[(size_t)k * ld]);
  out[t] = m;
}

// z -> output copy + f16 copy for decoder GEMM A
__global__ void z_finish_kernel(const float* __restrict__ z32, float* __restrict__ zOut,
                                _Float16* __restrict__ zh)
{
  int t = blockIdx.x * blockDim.x + threadIdx.x;
  if (t >= 64 * 1024) return;
  float v = z32[t];
  zOut[t] = v;
  zh[t] = (_Float16)v;
}

// grid@Wg^T : gp[p,o] = gx*W[o,1024] + gy*W[o,1025]  (46x46 linspace grid)
__global__ void gridpart_kernel(const float* __restrict__ W0, float* __restrict__ gp)
{
  int t = blockIdx.x * blockDim.x + threadIdx.x;
  if (t >= 2048 * 512) return;
  int p = t >> 9, o = t & 511;
  int ix = p / 46, iy = p % 46;
  float gx = -1.f + (2.f / 45.f) * (float)ix;
  float gy = -1.f + (2.f / 45.f) * (float)iy;
  gp[t] = gx * W0[(size_t)o * 1026 + 1024] + gy * W0[(size_t)o * 1026 + 1025];
}

// h1[b*2048+p, o] = relu((zpart[b,o] + gp[p,o]) * sc[o] + sh[o])  -> f16
__global__ void combine_kernel(const float* __restrict__ zpart, const float* __restrict__ gp,
                               const float* __restrict__ sc, const float* __restrict__ sh,
                               _Float16* __restrict__ h1)
{
  int t = blockIdx.x * blockDim.x + threadIdx.x;       // 131072*512 = 2^26
  if (t >= (64*2048) * 512) return;
  int row = t >> 9, o = t & 511;
  int b = row >> 11, p = row & 2047;
  float y = (zpart[(size_t)b*512 + o] + gp[(size_t)p*512 + o]) * sc[o] + sh[o];
  h1[t] = (_Float16)fmaxf(y, 0.f);
}

// ===========================================================================
extern "C" void kernel_launch(void* const* d_in, const int* in_sizes, int n_in,
                              void* d_out, int out_size, void* d_ws, size_t ws_size,
                              hipStream_t stream) {
  (void)in_sizes; (void)n_in; (void)out_size; (void)ws_size;
  const int B = 64;
  const float* x = (const float*)d_in[0];

  // --- flattened param indices (insertion order, depth-first; bn=None / relu
  //     flags carry no tensor data; relu pattern hardcoded: T,T,T,F) ---
  static const int WI[14]    = {1,7,13, 19,25,31, 37,43,49,55, 61,67,73,75};
  static const int HASBN[14] = {1,1,1,  1,1,1,    1,1,1,1,     1,1,0,0};
  static const int COUT[14]  = {64,64,128, 128,128,256, 256,256,512,1024, 512,256,128,3};
  static const int CIN[14]   = {3,64,64,   131,128,128, 259,256,256,512,  1024,512,256,128};
  static const int SRCLD[14] = {3,64,64,   131,128,128, 259,256,256,512,  1026,512,256,128};
  // note l=10 (decoder L0): pack only the z-columns (cin=1024 of 1026)

  // --- bump allocator over d_ws ---
  char* ws = (char*)d_ws;
  size_t off = 0;
  auto alloc = [&](size_t bytes) -> void* {
    off = (off + 255) & ~(size_t)255;
    void* p = (void*)(ws + off);
    off += bytes;
    return p;
  };

  int*   fpsIdx1 = (int*)  alloc((size_t)B*64*4);
  float* nx1     = (float*)alloc((size_t)B*64*3*4);
  int*   idx1    = (int*)  alloc((size_t)B*64*32*4);
  int*   fpsIdx2 = (int*)  alloc((size_t)B*16*4);
  float* nx2     = (float*)alloc((size_t)B*16*3*4);
  int*   idx2    = (int*)  alloc((size_t)B*16*32*4);
  float* feats1  = (float*)alloc((size_t)B*64*128*4);
  float* feats2  = (float*)alloc((size_t)B*16*256*4);
  float* z32     = (float*)alloc((size_t)B*1024*4);
  _Float16* zh   = (_Float16*)alloc((size_t)B*1024*2);
  float* zpart   = (float*)alloc((size_t)B*512*4);
  float* gp      = (float*)alloc((size_t)2048*512*4);

  _Float16* Wh[14]; float* SC[14]; float* SH[14];
  int KPAD[14], CPAD[14];
  for (int l = 0; l < 14; ++l) {
    KPAD[l] = ((CIN[l]  + 31) / 32) * 32;
    CPAD[l] = ((COUT[l] + 15) / 16) * 16;
    Wh[l] = (_Float16*)alloc((size_t)CPAD[l] * KPAD[l] * 2);
    SC[l] = (float*)alloc((size_t)CPAD[l] * 4);
    SH[l] = (float*)alloc((size_t)CPAD[l] * 4);
  }
  _Float16* bufA = (_Float16*)alloc((size_t)131072 * 512 * 2);   // 128 MiB
  _Float16* bufB = (_Float16*)alloc((size_t)131072 * 256 * 2);   //  64 MiB

  // --- fold weights+BN to f16 / scale / shift ---
  for (int l = 0; l < 14; ++l) {
    const float* W  = (const float*)d_in[WI[l]];
    const float* bb = (const float*)d_in[WI[l] + 1];
    const float *ga = nullptr, *be = nullptr, *me = nullptr, *va = nullptr;
    if (HASBN[l]) {
      ga = (const float*)d_in[WI[l] + 2]; be = (const float*)d_in[WI[l] + 3];
      me = (const float*)d_in[WI[l] + 4]; va = (const float*)d_in[WI[l] + 5];
    }
    prep_layer<<<CPAD[l], 128, 0, stream>>>(W, bb, ga, be, me, va,
        COUT[l], CIN[l], SRCLD[l], KPAD[l], CPAD[l], Wh[l], SC[l], SH[l]);
  }

  // ------------------------------- SA1 -------------------------------------
  fps_kernel  <<<B, 256, 0, stream>>>(x, 2048, 64, fpsIdx1, nx1);
  ballq_kernel<<<B*64, 32, 0, stream>>>(x, nx1, 2048, 64, 32, 0.04f, idx1);
  group1_kernel<<<CDIV(131072,256), 256, 0, stream>>>(x, nx1, idx1, bufA);
  // 131072 rows: 3(->32)->64->64->128, 32x32 macro-tiles
  wmma_gemm2x2<<<CDIV(4096*2,8), 256, 0, stream>>>(bufA, 32,  Wh[0], SC[0], SH[0], 4096, 2, 64,  bufB, nullptr, 64, 1);
  wmma_gemm2x2<<<CDIV(4096*2,8), 256, 0, stream>>>(bufB, 64,  Wh[1], SC[1], SH[1], 4096, 2, 64,  bufA, nullptr, 64, 1);
  wmma_gemm2x2<<<CDIV(4096*4,8), 256, 0, stream>>>(bufA, 64,  Wh[2], SC[2], SH[2], 4096, 4, 128, bufB, nullptr, 128, 1);
  maxpool_kernel<<<CDIV(4096*128,256), 256, 0, stream>>>(bufB, 128, 32, 128, 4096, feats1);

  // ------------------------------- SA2 -------------------------------------
  fps_kernel  <<<B, 256, 0, stream>>>(nx1, 64, 16, fpsIdx2, nx2);
  ballq_kernel<<<B*16, 32, 0, stream>>>(nx1, nx2, 64, 16, 32, 0.16f, idx2);
  group2_kernel<<<CDIV(32768,256), 256, 0, stream>>>(nx1, nx2, feats1, idx2, bufA);
  wmma_gemm2x2<<<CDIV(1024*4,8), 256, 0, stream>>>(bufA, 160, Wh[3], SC[3], SH[3], 1024, 4, 128, bufB, nullptr, 128, 1);
  wmma_gemm2x2<<<CDIV(1024*4,8), 256, 0, stream>>>(bufB, 128, Wh[4], SC[4], SH[4], 1024, 4, 128, bufA, nullptr, 128, 1);
  wmma_gemm2x2<<<CDIV(1024*8,8), 256, 0, stream>>>(bufA, 128, Wh[5], SC[5], SH[5], 1024, 8, 256, bufB, nullptr, 256, 1);
  maxpool_kernel<<<CDIV(1024*256,256), 256, 0, stream>>>(bufB, 256, 32, 256, 1024, feats2);

  // ----------------------------- SA3 (global) ------------------------------
  group3_kernel<<<CDIV(1024,256), 256, 0, stream>>>(nx2, feats2, bufA);
  wmma_gemm2x2<<<CDIV(32*8,8),  256, 0, stream>>>(bufA, 288, Wh[6], SC[6], SH[6], 32, 8,  256,  bufB, nullptr, 256, 1);
  wmma_gemm2x2<<<CDIV(32*8,8),  256, 0, stream>>>(bufB, 256, Wh[7], SC[7], SH[7], 32, 8,  256,  bufA, nullptr, 256, 1);
  wmma_gemm2x2<<<CDIV(32*16,8), 256, 0, stream>>>(bufA, 256, Wh[8], SC[8], SH[8], 32, 16, 512,  bufB, nullptr, 512, 1);
  wmma_gemm2x2<<<CDIV(32*32,8), 256, 0, stream>>>(bufB, 512, Wh[9], SC[9], SH[9], 32, 32, 1024, bufA, nullptr, 1024, 1);
  maxpool_kernel<<<CDIV(64*1024,256), 256, 0, stream>>>(bufA, 1024, 16, 1024, 64, z32);

  // ------------------------------ decoder ----------------------------------
  float* recon = (float*)d_out;                 // (B,2048,3)
  float* zOut  = (float*)d_out + (size_t)B*2048*3;
  z_finish_kernel<<<CDIV(65536,256), 256, 0, stream>>>(z32, zOut, zh);
  // zpart = z @ Wz^T  (raw, 64x1024x512)
  wmma_gemm2x2<<<CDIV(2*16,8), 256, 0, stream>>>(zh, 1024, Wh[10], nullptr, nullptr, 2, 16, 512,
                                                 nullptr, zpart, 512, 0);
  gridpart_kernel<<<CDIV(2048*512,256), 256, 0, stream>>>((const float*)d_in[61], gp);
  combine_kernel<<<CDIV(131072*512,256), 256, 0, stream>>>(zpart, gp, SC[10], SH[10], bufA);
  // 131072 rows: 512->256 (BN,relu), 256->128 (relu), 128->3 (linear, fp32 out)
  wmma_gemm2x2<<<CDIV(4096*8,8), 256, 0, stream>>>(bufA, 512, Wh[11], SC[11], SH[11], 4096, 8, 256, bufB, nullptr, 256, 1);
  wmma_gemm2x2<<<CDIV(4096*4,8), 256, 0, stream>>>(bufB, 256, Wh[12], SC[12], SH[12], 4096, 4, 128, bufA, nullptr, 128, 1);
  wmma_gemm<<<CDIV(8192*1,8),    256, 0, stream>>>(bufA, 128, Wh[13], SC[13], SH[13], 8192, 1, 3,
                                                   nullptr, recon, 3, 0);
}
